// EfficientSelfAttention_32074815767319
// MI455X (gfx1250) — matrix operand order repaired and checked by reading, not compiled, not run on hardware
//
#include <hip/hip_runtime.h>
#include <hip/hip_bf16.h>

#define BATCH 8
#define NQ    16384
#define CDIM  64
#define NK    256

typedef __attribute__((ext_vector_type(16))) __bf16 v16bf;
typedef __attribute__((ext_vector_type(8)))  __bf16 v8bf;
typedef __attribute__((ext_vector_type(8)))  float  v8f;

union V16U { v16bf v; v8bf h[2]; __bf16 e[16]; };

// Compiler-only fence: LDS pipe is in-order per wave (DScnt, ISA §11), so
// wave-private cross-lane staging needs only protection from compiler reordering.
#define LDSF() asm volatile("" ::: "memory")

static __device__ __forceinline__ v8f wmma_bf16(v16bf a, v16bf b, v8f c) {
  return __builtin_amdgcn_wmma_f32_16x16x32_bf16(false, a, false, b, (short)0, c, false, false);
}

// A-frag (16x32 bf16, K-step s, lane half) from a contiguous f32 row.
// lane: row = lane%16; element e -> K = s*32 + half*16 + e
static __device__ __forceinline__ v16bf a_frag_f32(const float* __restrict__ rowp, int s, int half) {
  const int c0 = s * 32 + half * 16;
  const float4 f0 = *(const float4*)(rowp + c0);
  const float4 f1 = *(const float4*)(rowp + c0 + 4);
  const float4 f2 = *(const float4*)(rowp + c0 + 8);
  const float4 f3 = *(const float4*)(rowp + c0 + 12);
  V16U u;
  u.e[0] = (__bf16)f0.x; u.e[1] = (__bf16)f0.y; u.e[2]  = (__bf16)f0.z; u.e[3]  = (__bf16)f0.w;
  u.e[4] = (__bf16)f1.x; u.e[5] = (__bf16)f1.y; u.e[6]  = (__bf16)f1.z; u.e[7]  = (__bf16)f1.w;
  u.e[8] = (__bf16)f2.x; u.e[9] = (__bf16)f2.y; u.e[10] = (__bf16)f2.z; u.e[11] = (__bf16)f2.w;
  u.e[12]= (__bf16)f3.x; u.e[13]= (__bf16)f3.y; u.e[14] = (__bf16)f3.z; u.e[15] = (__bf16)f3.w;
  return u.v;
}

// A-frag from a bf16 row (two 16B loads)
static __device__ __forceinline__ v16bf a_frag_bf16(const __bf16* __restrict__ rowp, int s, int half) {
  V16U u;
  u.h[0] = *(const v8bf*)(rowp + s * 32 + half * 16);
  u.h[1] = *(const v8bf*)(rowp + s * 32 + half * 16 + 8);
  return u.v;
}

// B-frag from matrix stored [N][K] bf16: lane col n, K = s*32 + half*16 + e (contiguous 32B)
static __device__ __forceinline__ v16bf b_frag(const __bf16* __restrict__ mat, int ldk, int n, int s, int half) {
  return *(const v16bf*)(mat + (size_t)n * ldk + s * 32 + half * 16);
}

// ---------------- kernel 1: repack weights to bf16 (transposed to [N][K]) ----------------
__global__ __launch_bounds__(256) void prep_weights(
    const float* __restrict__ Wq, const float* __restrict__ Wkv,
    const float* __restrict__ Wp, const float* __restrict__ srw,
    __bf16* __restrict__ WqT, __bf16* __restrict__ WkvT,
    __bf16* __restrict__ WpT, __bf16* __restrict__ Wct)
{
  const int i = blockIdx.x * 256 + threadIdx.x;
  if (i < 4096) WqT[i] = (__bf16)Wq[(i & 63) * 64 + (i >> 6)];    // WqT[o][c] = Wq[c][o]
  if (i < 8192) WkvT[i] = (__bf16)Wkv[(i & 63) * 128 + (i >> 6)]; // WkvT[o][c] = Wkv[c][o], o<128
  if (i < 4096) WpT[i] = (__bf16)Wp[(i & 63) * 64 + (i >> 6)];
  if (i < 262144) {                                               // Wct[kk][o][c] = sr_w[o][c][ky][kx]
    const int kk = i >> 12, o = (i >> 6) & 63, c = i & 63;
    Wct[i] = (__bf16)srw[o * 4096 + c * 64 + kk];
  }
}

// ---------------- kernel 2a: SR conv partial sums (split over 4 K-groups) ----------------
// grid: 512 blocks of 1 wave: blk = b*64 + ptile*4 + g ; each wave: 16 pixels, 16 (ky,kx) slices
__global__ __launch_bounds__(32) void conv_part(
    const float* __restrict__ x, const __bf16* __restrict__ Wct,
    float* __restrict__ xr_part)    // [4][8][256][64] f32
{
  const int lane = threadIdx.x & 31;
  const int half = lane >> 4, lq = lane & 15;
  const int g = blockIdx.x & 3;
  const int ptile = (blockIdx.x >> 2) & 15;     // 16 pixels per tile, oy = ptile, ox = lq
  const int b = blockIdx.x >> 6;
  const int oy = ptile;

  v8f acc[4];
#pragma unroll
  for (int t = 0; t < 4; ++t) for (int j = 0; j < 8; ++j) acc[t][j] = 0.f;

  for (int kk = g * 16; kk < g * 16 + 16; ++kk) {
    const int ky = kk >> 3, kx = kk & 7;
    const int xrow = (oy * 8 + ky) * 128 + lq * 8 + kx;
    const float* rowp = x + ((size_t)b * NQ + xrow) * CDIM;
    const __bf16* wkk = Wct + (size_t)kk * 4096;
#pragma unroll
    for (int s = 0; s < 2; ++s) {
      v16bf af = a_frag_f32(rowp, s, half);
#pragma unroll
      for (int t = 0; t < 4; ++t)
        acc[t] = wmma_bf16(af, b_frag(wkk, 64, t * 16 + lq, s, half), acc[t]);
    }
  }
  float* dst = xr_part + (((size_t)g * BATCH + b) * NK + ptile * 16) * CDIM;
#pragma unroll
  for (int t = 0; t < 4; ++t)
#pragma unroll
    for (int j = 0; j < 8; ++j)
      dst[(j + 8 * half) * CDIM + t * 16 + lq] = acc[t][j];
}

// ---------------- kernel 2b: sum partials + conv bias + LN + KV projection ----------------
__global__ __launch_bounds__(256) void ln_kv(
    const float* __restrict__ xr_part, const float* __restrict__ srb,
    const float* __restrict__ lng, const float* __restrict__ lnb,
    const float* __restrict__ bkv, const __bf16* __restrict__ WkvT,
    __bf16* __restrict__ Kbf, __bf16* __restrict__ Vt)
{
  __shared__ __attribute__((aligned(32))) __bf16 sStage[8][16 * CDIM];
  const int w = threadIdx.x >> 5, lane = threadIdx.x & 31;
  const int half = lane >> 4, lq = lane & 15;
  const int b = blockIdx.x >> 1;
  const int pbase = (blockIdx.x & 1) * 128 + w * 16;

  // gather the 4 K-group partials into D-frag layout
  v8f acc[4];
#pragma unroll
  for (int t = 0; t < 4; ++t) {
#pragma unroll
    for (int j = 0; j < 8; ++j) {
      const size_t idx = ((size_t)b * NK + pbase + j + 8 * half) * CDIM + t * 16 + lq;
      float v2 = xr_part[idx];
      v2 += xr_part[(size_t)1 * BATCH * NK * CDIM + idx];
      v2 += xr_part[(size_t)2 * BATCH * NK * CDIM + idx];
      v2 += xr_part[(size_t)3 * BATCH * NK * CDIM + idx];
      acc[t][j] = v2;
    }
  }
  // conv bias + LayerNorm stats per pixel row (m = j + 8*half)
  float sum[8], sq[8];
#pragma unroll
  for (int j = 0; j < 8; ++j) { sum[j] = 0.f; sq[j] = 0.f; }
#pragma unroll
  for (int t = 0; t < 4; ++t) {
    const float bb = srb[t * 16 + lq];
#pragma unroll
    for (int j = 0; j < 8; ++j) {
      float v2 = acc[t][j] + bb; acc[t][j] = v2; sum[j] += v2; sq[j] += v2 * v2;
    }
  }
#pragma unroll
  for (int m = 1; m < 16; m <<= 1)
#pragma unroll
    for (int j = 0; j < 8; ++j) { sum[j] += __shfl_xor(sum[j], m, 32); sq[j] += __shfl_xor(sq[j], m, 32); }
  float mu[8], rstd[8];
#pragma unroll
  for (int j = 0; j < 8; ++j) {
    mu[j] = sum[j] * (1.f / 64.f);
    const float var = sq[j] * (1.f / 64.f) - mu[j] * mu[j];
    rstd[j] = rsqrtf(var + 1e-5f);
  }
#pragma unroll
  for (int t = 0; t < 4; ++t) {
    const int n = t * 16 + lq;
    const float gg = lng[n], be = lnb[n];
#pragma unroll
    for (int j = 0; j < 8; ++j)
      sStage[w][(j + 8 * half) * CDIM + n] = (__bf16)((acc[t][j] - mu[j]) * rstd[j] * gg + be);
  }
  LDSF();
  const v16bf ra0 = a_frag_bf16(&sStage[w][lq * CDIM], 0, half);
  const v16bf ra1 = a_frag_bf16(&sStage[w][lq * CDIM], 1, half);
#pragma unroll
  for (int t = 0; t < 8; ++t) {
    v8f kv;
#pragma unroll
    for (int j = 0; j < 8; ++j) kv[j] = 0.f;
    kv = wmma_bf16(ra0, b_frag(WkvT, 64, t * 16 + lq, 0, half), kv);
    kv = wmma_bf16(ra1, b_frag(WkvT, 64, t * 16 + lq, 1, half), kv);
    const int n = t * 16 + lq;
    const float bb = bkv[n];
#pragma unroll
    for (int j = 0; j < 8; ++j) {
      const float v2 = kv[j] + bb;
      const int p = pbase + j + 8 * half;
      if (t < 4) Kbf[((size_t)b * NK + p) * CDIM + n] = (__bf16)v2;
      else       Vt[((size_t)b * CDIM + (n - 64)) * NK + p] = (__bf16)v2;
    }
  }
}

// ---------------- kernel 3: fused Q-proj + attention(2-pass softmax) + out-proj ----------------
__global__ __launch_bounds__(256) void sra_attn(
    const float* __restrict__ x, const float* __restrict__ bq, const float* __restrict__ bp,
    const __bf16* __restrict__ WqTg, const __bf16* __restrict__ WpT,
    const __bf16* __restrict__ Kg, const __bf16* __restrict__ Vt,
    float* __restrict__ out)
{
  __shared__ __attribute__((aligned(32))) __bf16 sK[NK * CDIM];        // 32 KB [key][d]
  __shared__ __attribute__((aligned(32))) __bf16 sWq[CDIM * CDIM];     // 8 KB [out][in]
  __shared__ __attribute__((aligned(32))) __bf16 sStage[8][16 * CDIM]; // 16 KB per-wave staging
  const int w = threadIdx.x >> 5, lane = threadIdx.x & 31;
  const int half = lane >> 4, lq = lane & 15;
  const int b = blockIdx.x >> 7;
  const int qbase = (blockIdx.x & 127) * 128 + w * 16;

  { // stage K (this batch) and WqT into LDS (cross-wave -> real barrier below)
    const __bf16* kb = Kg + (size_t)b * NK * CDIM;
    for (int i = threadIdx.x * 8; i < NK * CDIM; i += 256 * 8)
      *(v8bf*)&sK[i] = *(const v8bf*)&kb[i];
    for (int i = threadIdx.x * 8; i < CDIM * CDIM; i += 256 * 8)
      *(v8bf*)&sWq[i] = *(const v8bf*)&WqTg[i];
  }
  __syncthreads();

  // Q = X @ Wq + bq (scatter D-frags to wave-private LDS, reload in A layout)
  const float* xrow = x + ((size_t)b * NQ + qbase + lq) * CDIM;
  const v16bf xa0 = a_frag_f32(xrow, 0, half);
  const v16bf xa1 = a_frag_f32(xrow, 1, half);
#pragma unroll
  for (int t = 0; t < 4; ++t) {
    v8f q;
#pragma unroll
    for (int j = 0; j < 8; ++j) q[j] = 0.f;
    q = wmma_bf16(xa0, b_frag(sWq, 64, t * 16 + lq, 0, half), q);
    q = wmma_bf16(xa1, b_frag(sWq, 64, t * 16 + lq, 1, half), q);
    const float bb = bq[t * 16 + lq];
#pragma unroll
    for (int j = 0; j < 8; ++j)
      sStage[w][(j + 8 * half) * CDIM + t * 16 + lq] = (__bf16)(q[j] + bb);
  }
  LDSF();
  const v16bf qa0 = a_frag_bf16(&sStage[w][lq * CDIM], 0, half);
  const v16bf qa1 = a_frag_bf16(&sStage[w][lq * CDIM], 1, half);
  LDSF();
  const float cs = 0.125f * 1.44269504088896340736f; // d^-0.5 * log2(e)

  // pass 1: row maxima of S = Q K^T
  float rm[8];
#pragma unroll
  for (int j = 0; j < 8; ++j) rm[j] = -3.0e38f;
  for (int ct = 0; ct < 16; ++ct) {
    v8f s2;
#pragma unroll
    for (int j = 0; j < 8; ++j) s2[j] = 0.f;
    s2 = wmma_bf16(qa0, b_frag(sK, 64, ct * 16 + lq, 0, half), s2);
    s2 = wmma_bf16(qa1, b_frag(sK, 64, ct * 16 + lq, 1, half), s2);
#pragma unroll
    for (int j = 0; j < 8; ++j) rm[j] = fmaxf(rm[j], s2[j]);
  }
#pragma unroll
  for (int m = 1; m < 16; m <<= 1)
#pragma unroll
    for (int j = 0; j < 8; ++j) rm[j] = fmaxf(rm[j], __shfl_xor(rm[j], m, 32));
#pragma unroll
  for (int j = 0; j < 8; ++j) rm[j] *= cs;   // fold scale*log2e into the max

  // pass 2: P = exp2(S*cs - rm), rowsum, O += P @ V
  float rs[8];
#pragma unroll
  for (int j = 0; j < 8; ++j) rs[j] = 0.f;
  v8f oacc[4];
#pragma unroll
  for (int t = 0; t < 4; ++t) for (int j = 0; j < 8; ++j) oacc[t][j] = 0.f;
  const __bf16* vtb = Vt + (size_t)b * CDIM * NK;
  for (int ck = 0; ck < 4; ++ck) {           // 4 key chunks of 64
#pragma unroll
    for (int t = 0; t < 4; ++t) {
      const int ct = ck * 4 + t;
      v8f s2;
#pragma unroll
      for (int j = 0; j < 8; ++j) s2[j] = 0.f;
      s2 = wmma_bf16(qa0, b_frag(sK, 64, ct * 16 + lq, 0, half), s2);
      s2 = wmma_bf16(qa1, b_frag(sK, 64, ct * 16 + lq, 1, half), s2);
#pragma unroll
      for (int j = 0; j < 8; ++j) {
        const float p = exp2f(fmaf(s2[j], cs, -rm[j]));
        rs[j] += p;
        sStage[w][(j + 8 * half) * CDIM + t * 16 + lq] = (__bf16)p;
      }
    }
    LDSF();
    const v16bf pa0 = a_frag_bf16(&sStage[w][lq * CDIM], 0, half);
    const v16bf pa1 = a_frag_bf16(&sStage[w][lq * CDIM], 1, half);
    LDSF();
#pragma unroll
    for (int t = 0; t < 4; ++t) {            // B = V^T rows [d][key], K-range ck*64..
      const __bf16* vrow = vtb + (size_t)(t * 16 + lq) * NK + ck * 64;
      oacc[t] = wmma_bf16(pa0, *(const v16bf*)(vrow + half * 16), oacc[t]);
      oacc[t] = wmma_bf16(pa1, *(const v16bf*)(vrow + 32 + half * 16), oacc[t]);
    }
  }
#pragma unroll
  for (int m = 1; m < 16; m <<= 1)
#pragma unroll
    for (int j = 0; j < 8; ++j) rs[j] += __shfl_xor(rs[j], m, 32);
  float inv[8];
#pragma unroll
  for (int j = 0; j < 8; ++j) inv[j] = 1.0f / rs[j];

  // normalize O, stage, then final = O @ Wp + bp
#pragma unroll
  for (int t = 0; t < 4; ++t)
#pragma unroll
    for (int j = 0; j < 8; ++j)
      sStage[w][(j + 8 * half) * CDIM + t * 16 + lq] = (__bf16)(oacc[t][j] * inv[j]);
  LDSF();
  const v16bf oa0 = a_frag_bf16(&sStage[w][lq * CDIM], 0, half);
  const v16bf oa1 = a_frag_bf16(&sStage[w][lq * CDIM], 1, half);
#pragma unroll
  for (int t = 0; t < 4; ++t) {
    v8f f;
#pragma unroll
    for (int j = 0; j < 8; ++j) f[j] = 0.f;
    f = wmma_bf16(oa0, b_frag(WpT, 64, t * 16 + lq, 0, half), f);
    f = wmma_bf16(oa1, b_frag(WpT, 64, t * 16 + lq, 1, half), f);
    const float bb = bp[t * 16 + lq];
#pragma unroll
    for (int j = 0; j < 8; ++j)
      out[((size_t)b * NQ + qbase + j + 8 * half) * CDIM + t * 16 + lq] = f[j] + bb;
  }
}

extern "C" void kernel_launch(void* const* d_in, const int* in_sizes, int n_in,
                              void* d_out, int out_size, void* d_ws, size_t ws_size,
                              hipStream_t stream) {
  const float* x   = (const float*)d_in[0];
  const float* Wq  = (const float*)d_in[3];
  const float* bq  = (const float*)d_in[4];
  const float* Wkv = (const float*)d_in[5];
  const float* bkv = (const float*)d_in[6];
  const float* srw = (const float*)d_in[7];
  const float* srb = (const float*)d_in[8];
  const float* lng = (const float*)d_in[9];
  const float* lnb = (const float*)d_in[10];
  const float* Wp  = (const float*)d_in[11];
  const float* bp  = (const float*)d_in[12];
  float* out = (float*)d_out;

  char* ws = (char*)d_ws;
  __bf16* WqT   = (__bf16*)(ws + 0);          // 8 KB
  __bf16* WpT   = (__bf16*)(ws + 8192);       // 8 KB
  __bf16* WkvT  = (__bf16*)(ws + 16384);      // 16 KB
  __bf16* Wct   = (__bf16*)(ws + 32768);      // 512 KB [kk][o][c]
  __bf16* Kbf   = (__bf16*)(ws + 557056);     // 256 KB [b][key][d]
  __bf16* Vt    = (__bf16*)(ws + 819200);     // 256 KB [b][d][key]
  float*  xrp   = (float*)(ws + 1081344);     // 2 MB   [4][8][256][64] conv partials

  prep_weights<<<1024, 256, 0, stream>>>(Wq, Wkv, Wp, srw, WqT, WkvT, WpT, Wct);
  conv_part<<<BATCH * 16 * 4, 32, 0, stream>>>(x, Wct, xrp);
  ln_kv<<<BATCH * 2, 256, 0, stream>>>(xrp, srb, lng, lnb, bkv, WkvT, Kbf, Vt);
  sra_attn<<<BATCH * 128, 256, 0, stream>>>(x, bq, bp, WqT, WpT, Kbf, Vt, out);
}
